// LinearAttention_26980984553882
// MI455X (gfx1250) — matrix operand order repaired and checked by reading, not compiled, not run on hardware
//
#include <hip/hip_runtime.h>
#include <hip/hip_bf16.h>

// ---------------------------------------------------------------------------
// LinearAttention block for MI455X (gfx1250), bf16 WMMA path, fp32 accumulate.
// x:(32,128,64,64) -> qkv proj -> q/k softmax -> context -> out proj -> GN.
// GEMMs: register double-buffered K-pipeline, 32x64 output tile per wave.
// context GEMM stages its transpose tile with global_load_async_to_lds_b128.
// ---------------------------------------------------------------------------

typedef __attribute__((ext_vector_type(16))) __bf16 v16bf;
typedef __attribute__((ext_vector_type(8)))  float  v8f;
typedef int v4i_vec __attribute__((vector_size(16)));   // matches builtin param

#define NPOS 4096   // 64*64
#define NB   32     // batch
#define CIN  128

#if __has_builtin(__builtin_amdgcn_global_load_async_to_lds_b128) && \
    __has_builtin(__builtin_amdgcn_s_wait_asynccnt)
#define HAVE_ASYNC_LDS 1
#endif

__device__ __forceinline__ unsigned short f2bf(float f) {
  union { float f; unsigned int u; } c; c.f = f;
  unsigned int u = c.u;
  unsigned int r = ((u >> 16) & 1u) + 0x7FFFu;   // round-to-nearest-even
  return (unsigned short)((u + r) >> 16);
}
__device__ __forceinline__ float bf2f(unsigned short h) {
  union { unsigned int u; float f; } c; c.u = ((unsigned int)h) << 16;
  return c.f;
}

union Frag {
  v16bf v;
  uint4 q[2];
  unsigned short us[16];
};

// A operand (16x32, M x K, row-major, K contiguous). ISA layout:
// lane = m + 16*hi holds K = {8hi..8hi+7} U {16+8hi..16+8hi+7} of row m.
__device__ __forceinline__ v16bf load_fragA(const unsigned short* base, int ld, int lane) {
  int m = lane & 15, hi = lane >> 4;
  const unsigned short* p = base + (size_t)m * ld + 8 * hi;
  Frag f;
  f.q[0] = *(const uint4*)(p);        // K = 8hi .. 8hi+7
  f.q[1] = *(const uint4*)(p + 16);   // K = 16+8hi .. 16+8hi+7
  return f.v;
}

// B operand (32x16, K x N, row-major, N contiguous): lane = n + 16*hi holds
// column n, K = {8hi..8hi+7} U {16+8hi..16+8hi+7} (same order as A).
__device__ __forceinline__ v16bf load_fragB(const unsigned short* base, int ld, int lane) {
  int n = lane & 15, hi = lane >> 4;
  Frag f;
#pragma unroll
  for (int s = 0; s < 16; ++s) {
    int k = (s < 8) ? (8 * hi + s) : (16 + 8 * hi + (s - 8));
    f.us[s] = base[(size_t)k * ld + n];
  }
  return f.v;
}

__device__ __forceinline__ v8f wmma_bf16(v16bf a, v16bf b, v8f c) {
  return __builtin_amdgcn_wmma_f32_16x16x32_bf16(false, a, false, b, (short)0, c, false, false);
}

// C/D layout: element r of lane L -> row = r + 8*(L>>4), col = L&15.
__device__ __forceinline__ void store_tile_bf16(unsigned short* base, int ld, int lane, v8f acc) {
  int col = lane & 15, hi = lane >> 4;
#pragma unroll
  for (int r = 0; r < 8; ++r)
    base[(size_t)(r + 8 * hi) * ld + col] = f2bf(acc[r]);
}

// ---------------------------------------------------------------------------
// Vectorized fp32 -> bf16 conversion (float4 in, 2x dword packed out).
__global__ void cvt4_f32_bf16(const float* __restrict__ src, unsigned short* __restrict__ dst, int n4) {
  int i = blockIdx.x * blockDim.x + threadIdx.x;
  if (i >= n4) return;
  float4 v = ((const float4*)src)[i];
  unsigned int lo = (unsigned int)f2bf(v.x) | ((unsigned int)f2bf(v.y) << 16);
  unsigned int hi = (unsigned int)f2bf(v.z) | ((unsigned int)f2bf(v.w) << 16);
  ((uint2*)dst)[i] = make_uint2(lo, hi);
}

// QKV projection: per batch OUT(384x4096) = Wqkv(384x128) * X(128x4096).
// grid (4096/256, 384/64, 32), block 256 (8 waves); wave = 32x64 of output.
// Register double-buffered over the 4 K-steps.
__global__ void qkv_gemm(const unsigned short* __restrict__ XBF,
                         const unsigned short* __restrict__ WQ,
                         unsigned short* __restrict__ QKV) {
  int b = blockIdx.z;
  int n0 = blockIdx.x * 256;
  int m0 = blockIdx.y * 64;
  int wave = threadIdx.x >> 5, lane = threadIdx.x & 31;
  int mbase = m0 + (wave & 1) * 32;
  int nbase = n0 + (wave >> 1) * 64;
  const unsigned short* X  = XBF + (size_t)b * CIN * NPOS;
  const unsigned short* W0 = WQ + (size_t)mbase * 128;
  v16bf a_cur[2], b_cur[4], a_nxt[2], b_nxt[4];
#pragma unroll
  for (int mt = 0; mt < 2; ++mt) a_cur[mt] = load_fragA(W0 + (size_t)mt * 16 * 128, 128, lane);
#pragma unroll
  for (int t = 0; t < 4; ++t) b_cur[t] = load_fragB(X + nbase + t * 16, NPOS, lane);
  v8f acc[2][4] = {};
#pragma unroll
  for (int k0 = 0; k0 < 128; k0 += 32) {
    bool more = (k0 + 32 < 128);
    if (more) {
#pragma unroll
      for (int mt = 0; mt < 2; ++mt)
        a_nxt[mt] = load_fragA(W0 + (size_t)mt * 16 * 128 + k0 + 32, 128, lane);
#pragma unroll
      for (int t = 0; t < 4; ++t)
        b_nxt[t] = load_fragB(X + (size_t)(k0 + 32) * NPOS + nbase + t * 16, NPOS, lane);
      if (k0 + 64 < 128)
        __builtin_prefetch(X + (size_t)(k0 + 64) * NPOS + nbase + (size_t)(lane & 3) * NPOS, 0, 1);
    }
#pragma unroll
    for (int mt = 0; mt < 2; ++mt)
#pragma unroll
      for (int t = 0; t < 4; ++t)
        acc[mt][t] = wmma_bf16(a_cur[mt], b_cur[t], acc[mt][t]);
    if (more) {
      a_cur[0] = a_nxt[0]; a_cur[1] = a_nxt[1];
#pragma unroll
      for (int t = 0; t < 4; ++t) b_cur[t] = b_nxt[t];
    }
  }
  unsigned short* out = QKV + (size_t)b * 384 * NPOS;
#pragma unroll
  for (int mt = 0; mt < 2; ++mt)
#pragma unroll
    for (int t = 0; t < 4; ++t)
      store_tile_bf16(out + (size_t)(mbase + mt * 16) * NPOS + nbase + t * 16, NPOS, lane, acc[mt][t]);
}

// Softmax of q over D=32 (channels 0..127 = h*32+d), then * D^-0.5. In place.
__global__ void q_softmax(unsigned short* __restrict__ QKV) {
  int idx = blockIdx.x * blockDim.x + threadIdx.x;   // B*4*4096 threads
  int n = idx & 4095;
  int bh = idx >> 12;
  int h = bh & 3, b = bh >> 2;
  unsigned short* p = QKV + ((size_t)b * 384 + (size_t)h * 32) * NPOS + n;
  float v[32], mx = -3.4e38f;
#pragma unroll
  for (int d = 0; d < 32; ++d) { v[d] = bf2f(p[(size_t)d * NPOS]); mx = fmaxf(mx, v[d]); }
  float s = 0.f;
#pragma unroll
  for (int d = 0; d < 32; ++d) { v[d] = __expf(v[d] - mx); s += v[d]; }
  float r = 0.17677669529663687f / s;                // scale / sum
#pragma unroll
  for (int d = 0; d < 32; ++d) p[(size_t)d * NPOS] = f2bf(v[d] * r);
}

// Softmax of k over n=4096 (channels 128..255). One 256-thread block per row.
__global__ void k_softmax(unsigned short* __restrict__ QKV) {
  int row = blockIdx.x;                               // 0..4095
  int b = row >> 7, c = 128 + (row & 127);
  unsigned short* p = QKV + ((size_t)b * 384 + c) * NPOS;
  int tid = threadIdx.x;
  float v[16], mx = -3.4e38f;
#pragma unroll
  for (int j = 0; j < 16; ++j) { v[j] = bf2f(p[tid + j * 256]); mx = fmaxf(mx, v[j]); }
  __shared__ float red[256];
  red[tid] = mx; __syncthreads();
  for (int s = 128; s > 0; s >>= 1) { if (tid < s) red[tid] = fmaxf(red[tid], red[tid + s]); __syncthreads(); }
  mx = red[0]; __syncthreads();
  float sum = 0.f;
#pragma unroll
  for (int j = 0; j < 16; ++j) { v[j] = __expf(v[j] - mx); sum += v[j]; }
  red[tid] = sum; __syncthreads();
  for (int s = 128; s > 0; s >>= 1) { if (tid < s) red[tid] += red[tid + s]; __syncthreads(); }
  float inv = 1.f / red[0];
#pragma unroll
  for (int j = 0; j < 16; ++j) p[tid + j * 256] = f2bf(v[j] * inv);
}

// context[d][e] = sum_n k[d][n]*v[e][n]; stored TRANSPOSED as CTX[e][d] (bf16).
// One wave per (b,h); v-tile transposed through LDS (async copy when available).
__global__ void context_gemm(const unsigned short* __restrict__ QKV,
                             unsigned short* __restrict__ CTX) {
  int bh = blockIdx.x;
  int h = bh & 3, b = bh >> 2;
  const unsigned short* K = QKV + ((size_t)b * 384 + 128 + (size_t)h * 32) * NPOS;
  const unsigned short* V = QKV + ((size_t)b * 384 + 256 + (size_t)h * 32) * NPOS;
  int lane = threadIdx.x;
  __shared__ __align__(16) unsigned short vt[32][40];  // 80B row pitch (16B aligned)
  v8f acc[2][2] = {};
  for (int n0 = 0; n0 < NPOS; n0 += 32) {
    const unsigned short* vr = V + (size_t)lane * NPOS + n0;
#ifdef HAVE_ASYNC_LDS
    {
      __attribute__((address_space(1))) v4i_vec* gsrc =
          (__attribute__((address_space(1))) v4i_vec*)(vr);
      __attribute__((address_space(3))) v4i_vec* ldst =
          (__attribute__((address_space(3))) v4i_vec*)(&vt[lane][0]);
      __builtin_amdgcn_global_load_async_to_lds_b128(gsrc,     ldst,     0, 0);
      __builtin_amdgcn_global_load_async_to_lds_b128(gsrc + 1, ldst + 1, 0, 0);
      __builtin_amdgcn_global_load_async_to_lds_b128(gsrc + 2, ldst + 2, 0, 0);
      __builtin_amdgcn_global_load_async_to_lds_b128(gsrc + 3, ldst + 3, 0, 0);
      __builtin_amdgcn_s_wait_asynccnt(0);
    }
#else
    *(uint4*)&vt[lane][0]  = *(const uint4*)(vr);
    *(uint4*)&vt[lane][8]  = *(const uint4*)(vr + 8);
    *(uint4*)&vt[lane][16] = *(const uint4*)(vr + 16);
    *(uint4*)&vt[lane][24] = *(const uint4*)(vr + 24);
#endif
    __syncthreads();
    Frag f0, f1;                                      // B(n,e) = v[e][n]
#pragma unroll
    for (int j = 0; j < 16; ++j) { int kk = (j < 8) ? (8 * (lane >> 4) + j) : (16 + 8 * (lane >> 4) + (j - 8)); f0.us[j] = vt[lane & 15][kk]; f1.us[j] = vt[16 + (lane & 15)][kk]; }
    v16bf a0 = load_fragA(K + n0, NPOS, lane);        // d rows 0..15
    v16bf a1 = load_fragA(K + (size_t)16 * NPOS + n0, NPOS, lane); // d 16..31
    acc[0][0] = wmma_bf16(a0, f0.v, acc[0][0]);
    acc[0][1] = wmma_bf16(a0, f1.v, acc[0][1]);
    acc[1][0] = wmma_bf16(a1, f0.v, acc[1][0]);
    acc[1][1] = wmma_bf16(a1, f1.v, acc[1][1]);
    __syncthreads();
  }
  unsigned short* C = CTX + (size_t)bh * 1024;        // (e,d), d contiguous
  int col = lane & 15, hi = lane >> 4;
#pragma unroll
  for (int dt = 0; dt < 2; ++dt)
#pragma unroll
    for (int et = 0; et < 2; ++et)
#pragma unroll
      for (int r = 0; r < 8; ++r) {
        int d = dt * 16 + r + 8 * hi;
        int e = et * 16 + col;
        C[(size_t)e * 32 + d] = f2bf(acc[dt][et][r]);
      }
}

// out[e][n] = sum_d ctxT[e][d] * q[d][n]   (single K=32 WMMA step)
// grid (4096/256, 1, 128 bh), block 256.
__global__ void attn_gemm(const unsigned short* __restrict__ QKV,
                          const unsigned short* __restrict__ CTX,
                          unsigned short* __restrict__ ATTN) {
  int bh = blockIdx.z;
  int h = bh & 3, b = bh >> 2;
  const unsigned short* Q = QKV + ((size_t)b * 384 + (size_t)h * 32) * NPOS;
  const unsigned short* C = CTX + (size_t)bh * 1024;
  unsigned short* O = ATTN + ((size_t)b * 128 + (size_t)h * 32) * NPOS;
  int wave = threadIdx.x >> 5, lane = threadIdx.x & 31;
  int ebase = (wave & 1) * 16;
  int nbase = blockIdx.x * 256 + (wave >> 1) * 64;
  v16bf a = load_fragA(C + (size_t)ebase * 32, 32, lane);
#pragma unroll
  for (int t = 0; t < 4; ++t) {
    v16bf bb = load_fragB(Q + nbase + t * 16, NPOS, lane);
    v8f acc = {};
    acc = wmma_bf16(a, bb, acc);
    store_tile_bf16(O + (size_t)ebase * NPOS + nbase + t * 16, NPOS, lane, acc);
  }
}

// Output projection: Y(128x4096) = Wout(128x128)*ATTN + bias, fp32 out,
// plus per-block partial sum/sumsq for GroupNorm.
// grid (16, 2, 32), block 256; wave = 32x64 tile; double-buffered K.
__global__ void out_gemm(const unsigned short* __restrict__ ATTN,
                         const unsigned short* __restrict__ WO,
                         const float* __restrict__ b_out,
                         float* __restrict__ Y,
                         float* __restrict__ part) {
  int b = blockIdx.z;
  int n0 = blockIdx.x * 256;
  int m0 = blockIdx.y * 64;
  int wave = threadIdx.x >> 5, lane = threadIdx.x & 31;
  int mbase = m0 + (wave & 1) * 32;
  int nbase = n0 + (wave >> 1) * 64;
  const unsigned short* A  = ATTN + (size_t)b * 128 * NPOS;
  const unsigned short* W0 = WO + (size_t)mbase * 128;
  v16bf a_cur[2], b_cur[4], a_nxt[2], b_nxt[4];
#pragma unroll
  for (int mt = 0; mt < 2; ++mt) a_cur[mt] = load_fragA(W0 + (size_t)mt * 16 * 128, 128, lane);
#pragma unroll
  for (int t = 0; t < 4; ++t) b_cur[t] = load_fragB(A + nbase + t * 16, NPOS, lane);
  v8f acc[2][4] = {};
#pragma unroll
  for (int k0 = 0; k0 < 128; k0 += 32) {
    bool more = (k0 + 32 < 128);
    if (more) {
#pragma unroll
      for (int mt = 0; mt < 2; ++mt)
        a_nxt[mt] = load_fragA(W0 + (size_t)mt * 16 * 128 + k0 + 32, 128, lane);
#pragma unroll
      for (int t = 0; t < 4; ++t)
        b_nxt[t] = load_fragB(A + (size_t)(k0 + 32) * NPOS + nbase + t * 16, NPOS, lane);
      if (k0 + 64 < 128)
        __builtin_prefetch(A + (size_t)(k0 + 64) * NPOS + nbase + (size_t)(lane & 3) * NPOS, 0, 1);
    }
#pragma unroll
    for (int mt = 0; mt < 2; ++mt)
#pragma unroll
      for (int t = 0; t < 4; ++t)
        acc[mt][t] = wmma_bf16(a_cur[mt], b_cur[t], acc[mt][t]);
    if (more) {
      a_cur[0] = a_nxt[0]; a_cur[1] = a_nxt[1];
#pragma unroll
      for (int t = 0; t < 4; ++t) b_cur[t] = b_nxt[t];
    }
  }
  int col = lane & 15, hi = lane >> 4;
  float bias[2][8];
#pragma unroll
  for (int mt = 0; mt < 2; ++mt)
#pragma unroll
    for (int r = 0; r < 8; ++r) bias[mt][r] = b_out[mbase + mt * 16 + r + 8 * hi];
  float* Yb = Y + (size_t)b * 128 * NPOS;
  float ls = 0.f, lsq = 0.f;
#pragma unroll
  for (int mt = 0; mt < 2; ++mt)
#pragma unroll
    for (int t = 0; t < 4; ++t)
#pragma unroll
      for (int r = 0; r < 8; ++r) {
        float yv = acc[mt][t][r] + bias[mt][r];
        Yb[(size_t)(mbase + mt * 16 + r + 8 * hi) * NPOS + nbase + t * 16 + col] = yv;
        ls += yv; lsq += yv * yv;
      }
  __shared__ float s1[256], s2[256];
  int tid = threadIdx.x;
  s1[tid] = ls; s2[tid] = lsq; __syncthreads();
  for (int s = 128; s > 0; s >>= 1) {
    if (tid < s) { s1[tid] += s1[tid + s]; s2[tid] += s2[tid + s]; }
    __syncthreads();
  }
  if (tid == 0) {
    int pb = b * 32 + blockIdx.y * 16 + blockIdx.x;   // 32 blocks per batch
    part[2 * pb]     = s1[0];
    part[2 * pb + 1] = s2[0];
  }
}

// Per-batch mean / rsqrt(var+eps); fixed-order reduction -> deterministic.
__global__ void gn_stats(const float* __restrict__ part, float* __restrict__ stats) {
  int b = threadIdx.x;
  if (b >= NB) return;
  float s = 0.f, sq = 0.f;
  for (int i = 0; i < 32; ++i) { s += part[2 * (b * 32 + i)]; sq += part[2 * (b * 32 + i) + 1]; }
  const float invN = 1.0f / 524288.0f;               // 128*4096
  float mean = s * invN;
  float var = sq * invN - mean * mean;
  stats[2 * b] = mean;
  stats[2 * b + 1] = rsqrtf(var + 1e-5f);
}

// In-place normalize + gamma/beta. float4 over (B,128,4096).
__global__ void gn_apply(float* __restrict__ Y, const float* __restrict__ stats,
                         const float* __restrict__ gamma, const float* __restrict__ beta) {
  size_t i = (size_t)blockIdx.x * blockDim.x + threadIdx.x;  // float4 index
  size_t pos = i * 4;
  int b = (int)(pos >> 19);                          // 128*4096 = 2^19
  int c = (int)((pos >> 12) & 127);
  float mean = stats[2 * b], inv = stats[2 * b + 1];
  float g = gamma[c] * inv, be = beta[c];
  float4 y = ((float4*)Y)[i];
  y.x = (y.x - mean) * g + be;
  y.y = (y.y - mean) * g + be;
  y.z = (y.z - mean) * g + be;
  y.w = (y.w - mean) * g + be;
  ((float4*)Y)[i] = y;
}

// ---------------------------------------------------------------------------
extern "C" void kernel_launch(void* const* d_in, const int* in_sizes, int n_in,
                              void* d_out, int out_size, void* d_ws, size_t ws_size,
                              hipStream_t stream) {
  const float* x      = (const float*)d_in[0];
  const float* w_qkv  = (const float*)d_in[1];
  const float* w_out  = (const float*)d_in[2];
  const float* b_out  = (const float*)d_in[3];
  const float* gamma  = (const float*)d_in[4];
  const float* beta   = (const float*)d_in[5];
  float* Y = (float*)d_out;
  char* ws = (char*)d_ws;

  unsigned short* XBF   = (unsigned short*)(ws + 0);          //  32 MiB
  unsigned short* QKVBF = (unsigned short*)(ws + 33554432);   //  96 MiB
  unsigned short* ATTN  = (unsigned short*)(ws + 134217728);  //  32 MiB
  unsigned short* WQ    = (unsigned short*)(ws + 167772160);  //  96 KiB
  unsigned short* WO    = (unsigned short*)(ws + 167870464);  //  32 KiB
  unsigned short* CTX   = (unsigned short*)(ws + 167903232);  // 256 KiB
  float*          PART  = (float*)(ws + 168165376);           //   8 KiB
  float*          STATS = (float*)(ws + 168181760);           // 256 B

  cvt4_f32_bf16<<<16384, 256, 0, stream>>>(x,     XBF, 4194304);
  cvt4_f32_bf16<<<48,    256, 0, stream>>>(w_qkv, WQ,  12288);
  cvt4_f32_bf16<<<16,    256, 0, stream>>>(w_out, WO,  4096);

  qkv_gemm<<<dim3(16, 6, 32), 256, 0, stream>>>(XBF, WQ, QKVBF);
  q_softmax<<<2048, 256, 0, stream>>>(QKVBF);
  k_softmax<<<4096, 256, 0, stream>>>(QKVBF);
  context_gemm<<<128, 32, 0, stream>>>(QKVBF, CTX);
  attn_gemm<<<dim3(16, 1, 128), 256, 0, stream>>>(QKVBF, CTX, ATTN);
  out_gemm<<<dim3(16, 2, 32), 256, 0, stream>>>(ATTN, WO, b_out, Y, PART);
  gn_stats<<<1, 32, 0, stream>>>(PART, STATS);
  gn_apply<<<16384, 256, 0, stream>>>(Y, STATS, gamma, beta);
}
